// ResidualGraphEncoder_84456236909203
// MI455X (gfx1250) — matrix operand order repaired and checked by reading, not compiled, not run on hardware
//
#include <hip/hip_runtime.h>
#include <hip/hip_bf16.h>

typedef __attribute__((ext_vector_type(16))) __bf16 v16bf;
typedef __attribute__((ext_vector_type(8)))  __bf16 v8bf;
typedef __attribute__((ext_vector_type(8)))  float  v8f;
typedef __attribute__((ext_vector_type(4)))  int    v4i;

#define HDIM 256
#define EDIM 64

#if defined(__has_builtin)
#if __has_builtin(__builtin_amdgcn_global_load_async_to_lds_b128)
#define HAS_ASYNC_LDS 1
#endif
#endif
#ifndef HAS_ASYNC_LDS
#define HAS_ASYNC_LDS 0
#endif

// Cooperatively stage one 8192-element (16 KB) weight slab into LDS.
// 1024 chunks of 8 bf16 (16 B) distributed over nth threads.
__device__ __forceinline__ void stage_slab(const __bf16* __restrict__ g,
                                           __bf16* l, int t, int nth) {
#if HAS_ASYNC_LDS
    for (int c = t; c < 1024; c += nth) {
        __builtin_amdgcn_global_load_async_to_lds_b128(
            (v4i*)(g + c * 8), (v4i*)(l + c * 8), 0, 0);
    }
#else
    for (int c = t; c < 1024; c += nth) {
        v8bf v = *(const v8bf*)(g + c * 8);
        *(v8bf*)(l + c * 8) = v;
    }
#endif
}

__device__ __forceinline__ void wait_stage() {
#if HAS_ASYNC_LDS
    asm volatile("s_wait_asynccnt 0x0" ::: "memory");
#endif
}

// ---------------- utility kernels ----------------

__global__ void copy_f32(const float* __restrict__ in, float* __restrict__ out, long n) {
    long g = (long)blockIdx.x * blockDim.x + threadIdx.x;
    if (g < n) out[g] = in[g];
}

__global__ void zero_f32(float* __restrict__ p, long n) {
    long g = (long)blockIdx.x * blockDim.x + threadIdx.x;
    if (g < n) p[g] = 0.0f;
}

__global__ void conv_bf16(const float* __restrict__ in, __bf16* __restrict__ out, long n) {
    long g = (long)blockIdx.x * blockDim.x + threadIdx.x;
    if (g < n) out[g] = (__bf16)in[g];
}

__global__ void deg_count(const int* __restrict__ dst, float* __restrict__ deg, int E) {
    int g = blockIdx.x * blockDim.x + threadIdx.x;
    if (g < E)
        __hip_atomic_fetch_add(&deg[dst[g]], 1.0f, __ATOMIC_RELAXED, __HIP_MEMORY_SCOPE_AGENT);
}

// Pack W[K][Nout] (f32) into WMMA B tiles (32x16): lane=((kr>>3)&1)*16+nr, elem=(kr&7)+(kr>>4)*8.
__global__ void pack_w(const float* __restrict__ W, __bf16* __restrict__ P, int K, int Nout) {
    int g = blockIdx.x * blockDim.x + threadIdx.x;
    int total = K * Nout;
    if (g >= total) return;
    int n = g % Nout;
    int k = g / Nout;
    int kt = k >> 5, kr = k & 31;
    int nt = n >> 4, nr = n & 15;
    int Nt = Nout >> 4;
    int lane = (((kr >> 3) & 1) << 4) + nr;
    int e = (kr & 7) + ((kr >> 4) << 3);
    P[(((size_t)kt * Nt + nt) << 9) + (lane << 4) + e] = (__bf16)W[g];
}

// ---------------- layernorm kernels (one wave32 per node) ----------------

__global__ __launch_bounds__(128) void ln_node(const float* __restrict__ h,
                                               const float* __restrict__ g,
                                               const float* __restrict__ b,
                                               __bf16* __restrict__ hn,
                                               __bf16* __restrict__ ybf, int N) {
    int w = threadIdx.x >> 5, lane = threadIdx.x & 31;
    int node = blockIdx.x * 4 + w;
    if (node >= N) return;
    const float* x = h + (long)node * HDIM;
    float vals[8], s = 0.f, ss = 0.f;
#pragma unroll
    for (int j = 0; j < 8; ++j) {
        float t = x[lane + j * 32];
        vals[j] = t; s += t; ss += t * t;
    }
#pragma unroll
    for (int o = 16; o > 0; o >>= 1) {
        s  += __shfl_xor(s, o, 32);
        ss += __shfl_xor(ss, o, 32);
    }
    float mean = s * (1.f / HDIM);
    float var  = ss * (1.f / HDIM) - mean * mean;
    float inv  = rsqrtf(var + 1e-5f);
#pragma unroll
    for (int j = 0; j < 8; ++j) {
        int idx = lane + j * 32;
        float r = (vals[j] - mean) * inv * g[idx] + b[idx];
        __bf16 bb = (__bf16)r;
        hn[(long)node * HDIM + idx] = bb;
        ybf[(long)node * 512 + idx] = bb;
    }
}

__global__ __launch_bounds__(128) void ln_agg(const float* __restrict__ agg,
                                              const float* __restrict__ deg,
                                              const float* __restrict__ g,
                                              const float* __restrict__ b,
                                              __bf16* __restrict__ ybf, int N) {
    int w = threadIdx.x >> 5, lane = threadIdx.x & 31;
    int node = blockIdx.x * 4 + w;
    if (node >= N) return;
    float dinv = 1.0f / fmaxf(deg[node], 1.0f);
    const float* x = agg + (long)node * HDIM;
    float vals[8], s = 0.f, ss = 0.f;
#pragma unroll
    for (int j = 0; j < 8; ++j) {
        float t = x[lane + j * 32] * dinv;
        vals[j] = t; s += t; ss += t * t;
    }
#pragma unroll
    for (int o = 16; o > 0; o >>= 1) {
        s  += __shfl_xor(s, o, 32);
        ss += __shfl_xor(ss, o, 32);
    }
    float mean = s * (1.f / HDIM);
    float var  = ss * (1.f / HDIM) - mean * mean;
    float inv  = rsqrtf(var + 1e-5f);
#pragma unroll
    for (int j = 0; j < 8; ++j) {
        int idx = lane + j * 32;
        float r = (vals[j] - mean) * inv * g[idx] + b[idx];
        ybf[(long)node * 512 + 256 + idx] = (__bf16)r;
    }
}

// ---------------- edge MLP ----------------

__device__ __forceinline__ v8bf load_xrow(int c, const __bf16* __restrict__ hn,
                                          long sOff, long dOff,
                                          const __bf16* __restrict__ ea, long eOff) {
    const __bf16* p;
    if (c < 256)      p = hn + sOff + c;
    else if (c < 512) p = hn + dOff + (c - 256);
    else              p = ea + eOff + (c - 512);
    return *(const v8bf*)p;
}

__global__ __launch_bounds__(128) void edge_mlp(const __bf16* __restrict__ hn,
                                                const __bf16* __restrict__ ea,
                                                const int* __restrict__ src,
                                                const int* __restrict__ dst,
                                                const __bf16* __restrict__ W1p,
                                                const float* __restrict__ b1,
                                                const __bf16* __restrict__ W2p,
                                                const float* __restrict__ b2,
                                                float* __restrict__ agg, int E) {
    __shared__ __align__(16) __bf16 xpose[4][16 * 256];   // 32 KB
    __shared__ __align__(16) __bf16 bb[2][8192];          // 32 KB (double-buffered B slab)
    int t = threadIdx.x;
    int w = t >> 5, lane = t & 31;
    int tile = blockIdx.x * 4 + w;
    bool live = (tile * 16 < E);
    int e0 = live ? tile * 16 : 0;
    int m = lane & 15, hf = lane >> 4, nb = lane & 15;
    int eA = e0 + m;
    long sOff = (long)src[eA] * HDIM;
    long dOff = (long)dst[eA] * HDIM;
    long eOff = (long)eA * EDIM;

    // ---------- GEMM1: [16,576] @ [576,256] ----------
    stage_slab(W1p, bb[0], t, 128);
    wait_stage();
    __syncthreads();
    v8f acc[16] = {};
    for (int kc = 0; kc < 18; ++kc) {
        int cur = kc & 1;
        if (kc < 17) stage_slab(W1p + (size_t)(kc + 1) * 8192, bb[cur ^ 1], t, 128);
        int c0 = kc * 32 + hf * 8;
        v8bf ch0 = load_xrow(c0,      hn, sOff, dOff, ea, eOff);
        v8bf ch1 = load_xrow(c0 + 16, hn, sOff, dOff, ea, eOff);
        v16bf a = __builtin_shufflevector(ch0, ch1, 0,1,2,3,4,5,6,7,8,9,10,11,12,13,14,15);
#pragma unroll
        for (int nt = 0; nt < 16; ++nt) {
            v16bf bfrag = *(const v16bf*)&bb[cur][nt * 512 + lane * 16];
            acc[nt] = __builtin_amdgcn_wmma_f32_16x16x32_bf16(
                false, a, false, bfrag, (short)0, acc[nt], false, false);
        }
        wait_stage();
        __syncthreads();
    }
    // bias + silu, C-layout -> per-wave LDS transpose buffer
#pragma unroll
    for (int nt = 0; nt < 16; ++nt) {
        int n = nt * 16 + nb;
        float bias = b1[n];
#pragma unroll
        for (int v = 0; v < 8; ++v) {
            float x = acc[nt][v] + bias;
            float sl = x / (1.0f + __expf(-x));
            xpose[w][(hf * 8 + v) * 256 + n] = (__bf16)sl;
        }
    }
    // ---------- GEMM2: [16,256] @ [256,256] ----------
    stage_slab(W2p, bb[0], t, 128);
    wait_stage();
    __syncthreads();
    v8f acc2[16] = {};
    for (int kc = 0; kc < 8; ++kc) {
        int cur = kc & 1;
        if (kc < 7) stage_slab(W2p + (size_t)(kc + 1) * 8192, bb[cur ^ 1], t, 128);
        int c0 = kc * 32 + hf * 8;
        v8bf ch0 = *(const v8bf*)&xpose[w][m * 256 + c0];
        v8bf ch1 = *(const v8bf*)&xpose[w][m * 256 + c0 + 16];
        v16bf a = __builtin_shufflevector(ch0, ch1, 0,1,2,3,4,5,6,7,8,9,10,11,12,13,14,15);
#pragma unroll
        for (int nt = 0; nt < 16; ++nt) {
            v16bf bfrag = *(const v16bf*)&bb[cur][nt * 512 + lane * 16];
            acc2[nt] = __builtin_amdgcn_wmma_f32_16x16x32_bf16(
                false, a, false, bfrag, (short)0, acc2[nt], false, false);
        }
        wait_stage();
        __syncthreads();
    }
    // bias + atomic scatter
    if (live) {
        int dIdx[8];
#pragma unroll
        for (int v = 0; v < 8; ++v) dIdx[v] = dst[e0 + hf * 8 + v];
#pragma unroll
        for (int nt = 0; nt < 16; ++nt) {
            int n = nt * 16 + nb;
            float bias = b2[n];
#pragma unroll
            for (int v = 0; v < 8; ++v) {
                float val = acc2[nt][v] + bias;
                __hip_atomic_fetch_add(&agg[(long)dIdx[v] * HDIM + n], val,
                                       __ATOMIC_RELAXED, __HIP_MEMORY_SCOPE_AGENT);
            }
        }
    }
}

// ---------------- node MLP ----------------

__global__ __launch_bounds__(64) void node_mlp(const __bf16* __restrict__ ybf,
                                               const __bf16* __restrict__ W1p,
                                               const float* __restrict__ b1,
                                               const __bf16* __restrict__ W2p,
                                               const float* __restrict__ b2,
                                               float* __restrict__ h, int N) {
    __shared__ __align__(16) __bf16 xpose[2][16 * 512];   // 32 KB
    __shared__ __align__(16) __bf16 bb[2][8192];          // 32 KB
    int t = threadIdx.x;
    int w = t >> 5, lane = t & 31;
    int tile = blockIdx.x * 2 + w;
    bool live = (tile * 16 < N);
    int r0 = live ? tile * 16 : 0;
    int m = lane & 15, hf = lane >> 4, nb = lane & 15;
    const __bf16* y = ybf + (long)(r0 + m) * 512;

    // ---------- GEMM1: [16,512] @ [512,512] in two 256-wide halves ----------
    for (int nh = 0; nh < 2; ++nh) {
        stage_slab(W1p + (size_t)(nh * 16) * 512, bb[0], t, 64);
        wait_stage();
        __syncthreads();
        v8f acc[16] = {};
        for (int kc = 0; kc < 16; ++kc) {
            int cur = kc & 1;
            if (kc < 15)
                stage_slab(W1p + (size_t)((kc + 1) * 32 + nh * 16) * 512, bb[cur ^ 1], t, 64);
            int c0 = kc * 32 + hf * 8;
            v8bf ch0 = *(const v8bf*)(y + c0);
            v8bf ch1 = *(const v8bf*)(y + c0 + 16);
            v16bf a = __builtin_shufflevector(ch0, ch1, 0,1,2,3,4,5,6,7,8,9,10,11,12,13,14,15);
#pragma unroll
            for (int nt = 0; nt < 16; ++nt) {
                v16bf bfrag = *(const v16bf*)&bb[cur][nt * 512 + lane * 16];
                acc[nt] = __builtin_amdgcn_wmma_f32_16x16x32_bf16(
                    false, a, false, bfrag, (short)0, acc[nt], false, false);
            }
            wait_stage();
            __syncthreads();
        }
#pragma unroll
        for (int nt = 0; nt < 16; ++nt) {
            int n = nh * 256 + nt * 16 + nb;
            float bias = b1[n];
#pragma unroll
            for (int v = 0; v < 8; ++v) {
                float x = acc[nt][v] + bias;
                float sl = x / (1.0f + __expf(-x));
                xpose[w][(hf * 8 + v) * 512 + n] = (__bf16)sl;
            }
        }
    }
    // ---------- GEMM2: [16,512] @ [512,256] ----------
    stage_slab(W2p, bb[0], t, 64);
    wait_stage();
    __syncthreads();
    v8f acc2[16] = {};
    for (int kc = 0; kc < 16; ++kc) {
        int cur = kc & 1;
        if (kc < 15) stage_slab(W2p + (size_t)(kc + 1) * 8192, bb[cur ^ 1], t, 64);
        int c0 = kc * 32 + hf * 8;
        v8bf ch0 = *(const v8bf*)&xpose[w][m * 512 + c0];
        v8bf ch1 = *(const v8bf*)&xpose[w][m * 512 + c0 + 16];
        v16bf a = __builtin_shufflevector(ch0, ch1, 0,1,2,3,4,5,6,7,8,9,10,11,12,13,14,15);
#pragma unroll
        for (int nt = 0; nt < 16; ++nt) {
            v16bf bfrag = *(const v16bf*)&bb[cur][nt * 512 + lane * 16];
            acc2[nt] = __builtin_amdgcn_wmma_f32_16x16x32_bf16(
                false, a, false, bfrag, (short)0, acc2[nt], false, false);
        }
        wait_stage();
        __syncthreads();
    }
    if (live) {
#pragma unroll
        for (int nt = 0; nt < 16; ++nt) {
            int n = nt * 16 + nb;
            float bias = b2[n];
#pragma unroll
            for (int v = 0; v < 8; ++v) {
                long r = (long)(r0 + hf * 8 + v);
                h[r * HDIM + n] += acc2[nt][v] + bias;
            }
        }
    }
}

// ---------------- host launcher ----------------

static inline long ceil_div(long a, long b) { return (a + b - 1) / b; }

extern "C" void kernel_launch(void* const* d_in, const int* in_sizes, int n_in,
                              void* d_out, int out_size, void* d_ws, size_t ws_size,
                              hipStream_t stream) {
    const float* node_state = (const float*)d_in[0];
    const float* edge_attr  = (const float*)d_in[1];
    const float* nn_g = (const float*)d_in[2];
    const float* nn_b = (const float*)d_in[3];
    const float* mn_g = (const float*)d_in[4];
    const float* mn_b = (const float*)d_in[5];
    const float* eW1  = (const float*)d_in[6];
    const float* eb1  = (const float*)d_in[7];
    const float* eW2  = (const float*)d_in[8];
    const float* eb2  = (const float*)d_in[9];
    const float* nW1  = (const float*)d_in[10];
    const float* nb1  = (const float*)d_in[11];
    const float* nW2  = (const float*)d_in[12];
    const float* nb2  = (const float*)d_in[13];
    const int*   ei   = (const int*)d_in[14];

    const int N = in_sizes[0] / HDIM;
    const int E = in_sizes[1] / EDIM;
    const int S = in_sizes[2] / HDIM;
    const int* src = ei;
    const int* dst = ei + E;

    char* ws = (char*)d_ws;
    size_t off = 0;
    auto alloc = [&](size_t bytes) { void* p = ws + off; off = (off + bytes + 255) & ~(size_t)255; return p; };
    __bf16* hn_bf = (__bf16*)alloc((size_t)N * HDIM * 2);
    __bf16* ybf   = (__bf16*)alloc((size_t)N * 512 * 2);
    float*  agg   = (float*)alloc((size_t)N * HDIM * 4);
    float*  deg   = (float*)alloc((size_t)N * 4);
    __bf16* ea_bf = (__bf16*)alloc((size_t)E * EDIM * 2);
    __bf16* eW1p  = (__bf16*)alloc((size_t)S * 576 * 256 * 2);
    __bf16* eW2p  = (__bf16*)alloc((size_t)S * 256 * 256 * 2);
    __bf16* nW1p  = (__bf16*)alloc((size_t)S * 512 * 512 * 2);
    __bf16* nW2p  = (__bf16*)alloc((size_t)S * 512 * 256 * 2);
    (void)ws_size;

    float* h = (float*)d_out;

    copy_f32<<<ceil_div((long)N * HDIM, 256), 256, 0, stream>>>(node_state, h, (long)N * HDIM);
    conv_bf16<<<ceil_div((long)E * EDIM, 256), 256, 0, stream>>>(edge_attr, ea_bf, (long)E * EDIM);
    for (int s = 0; s < S; ++s) {
        pack_w<<<ceil_div(576 * 256, 256), 256, 0, stream>>>(eW1 + (size_t)s * 576 * 256, eW1p + (size_t)s * 576 * 256, 576, 256);
        pack_w<<<ceil_div(256 * 256, 256), 256, 0, stream>>>(eW2 + (size_t)s * 256 * 256, eW2p + (size_t)s * 256 * 256, 256, 256);
        pack_w<<<ceil_div(512 * 512, 256), 256, 0, stream>>>(nW1 + (size_t)s * 512 * 512, nW1p + (size_t)s * 512 * 512, 512, 512);
        pack_w<<<ceil_div(512 * 256, 256), 256, 0, stream>>>(nW2 + (size_t)s * 512 * 256, nW2p + (size_t)s * 512 * 256, 512, 256);
    }
    zero_f32<<<ceil_div((long)N, 256), 256, 0, stream>>>(deg, (long)N);
    deg_count<<<ceil_div((long)E, 256), 256, 0, stream>>>(dst, deg, E);

    const int Etiles = E / 16;
    const int Ntiles = N / 16;
    for (int s = 0; s < S; ++s) {
        ln_node<<<ceil_div((long)N, 4), 128, 0, stream>>>(h, nn_g + (size_t)s * HDIM, nn_b + (size_t)s * HDIM, hn_bf, ybf, N);
        zero_f32<<<ceil_div((long)N * HDIM, 256), 256, 0, stream>>>(agg, (long)N * HDIM);
        edge_mlp<<<ceil_div((long)Etiles, 4), 128, 0, stream>>>(
            hn_bf, ea_bf, src, dst,
            eW1p + (size_t)s * 576 * 256, eb1 + (size_t)s * HDIM,
            eW2p + (size_t)s * 256 * 256, eb2 + (size_t)s * HDIM, agg, E);
        ln_agg<<<ceil_div((long)N, 4), 128, 0, stream>>>(agg, deg, mn_g + (size_t)s * HDIM, mn_b + (size_t)s * HDIM, ybf, N);
        node_mlp<<<ceil_div((long)Ntiles, 2), 64, 0, stream>>>(
            ybf,
            nW1p + (size_t)s * 512 * 512, nb1 + (size_t)s * 512,
            nW2p + (size_t)s * 512 * 256, nb2 + (size_t)s * HDIM, h, N);
    }
}
